// HKANGNN_11536282157101
// MI455X (gfx1250) — matrix operand order repaired and checked by reading, not compiled, not run on hardware
//
#include <hip/hip_runtime.h>
#include <hip/hip_bf16.h>
#include <stdint.h>

#define H 256
#define NE 50000
#define NU 150000
#define NS 10000
#define N_UE 500000
#define N_SE 100000

typedef __attribute__((ext_vector_type(16))) __bf16 v16bf;
typedef __attribute__((ext_vector_type(8)))  float  v8f;
typedef __attribute__((ext_vector_type(4)))  unsigned int u32x4;
typedef __attribute__((ext_vector_type(8)))  int          i32x8;
typedef __attribute__((ext_vector_type(4)))  int          i32x4;

#if defined(__has_builtin)
#if __has_builtin(__builtin_amdgcn_tensor_load_to_lds) && __has_builtin(__builtin_amdgcn_s_wait_tensorcnt)
#define HAS_TDM 1
#endif
#endif

// ---------------------------------------------------------------------------
// helpers
// ---------------------------------------------------------------------------
__device__ __forceinline__ unsigned short f32_to_bf16_rne(float f) {
    unsigned int u = __float_as_uint(f);
    unsigned int r = u + 0x7FFFu + ((u >> 16) & 1u);
    return (unsigned short)(r >> 16);
}

union FragU {
    v16bf v;
    uint4 q[2];
    unsigned short s[16];
};

// ---------------------------------------------------------------------------
// Zero-fill / f32 -> bf16 conversion (grid-stride, graph-safe)
// ---------------------------------------------------------------------------
__global__ void zero_f32(float* __restrict__ p, size_t n) {
    size_t i = (size_t)blockIdx.x * blockDim.x + threadIdx.x;
    size_t stride = (size_t)gridDim.x * blockDim.x;
    for (; i < n; i += stride) p[i] = 0.0f;
}

__global__ void cvt_f32_bf16(const float* __restrict__ in,
                             unsigned short* __restrict__ out, size_t n) {
    size_t i = (size_t)blockIdx.x * blockDim.x + threadIdx.x;
    size_t stride = (size_t)gridDim.x * blockDim.x;
    for (; i < n; i += stride) out[i] = f32_to_bf16_rne(in[i]);
}

// ---------------------------------------------------------------------------
// TDM: issue one 2D bf16 tile load (tile_rows x tile_cols) into LDS.
// Tensor dims are given RELATIVE to the tile start so the TDM's OOB
// zero-fill handles ragged M edges for free. data_size = 2 bytes.
// 6-arg builtin variant (clang-23 / therock-10.0 headers).
// ---------------------------------------------------------------------------
#ifdef HAS_TDM
__device__ __forceinline__ void tdm_load_tile_bf16(
    const unsigned short* gbase, unsigned lds_off,
    int row0, int rows_total, int col0, int row_stride_elems,
    int tile_rows, int tile_cols)
{
    uint64_t gaddr = (uint64_t)(uintptr_t)gbase +
                     ((uint64_t)row0 * (uint64_t)row_stride_elems + (uint64_t)col0) * 2ull;
    unsigned td0 = (unsigned)(row_stride_elems - col0);   // remaining cols
    unsigned td1 = (unsigned)(rows_total - row0);         // remaining rows
    unsigned long long st0 = (unsigned long long)row_stride_elems;

    u32x4 g0;
    g0[0] = 1u;                                            // count=1, user mode
    g0[1] = lds_off;                                       // LDS byte address
    g0[2] = (unsigned)(gaddr & 0xFFFFFFFFu);               // global_addr[31:0]
    g0[3] = (unsigned)((gaddr >> 32) & 0x1FFFFFFu) | (2u << 30); // [56:32] | type=2

    i32x8 g1;
    g1[0] = (int)(1u << 16);                               // data_size = 2B
    g1[1] = (int)((td0 & 0xFFFFu) << 16);                  // tensor_dim0[15:0]
    g1[2] = (int)((td0 >> 16) | ((td1 & 0xFFFFu) << 16));  // dim0 hi | dim1 lo
    g1[3] = (int)((td1 >> 16) | ((unsigned)tile_cols << 16)); // dim1 hi | tile_dim0
    g1[4] = (int)((unsigned)tile_rows & 0xFFFFu);          // tile_dim1 (tile_dim2=0)
    g1[5] = (int)(st0 & 0xFFFFFFFFu);                      // dim0_stride[31:0]
    g1[6] = (int)((st0 >> 32) & 0xFFFFu);                  // dim0_stride[47:32]
    g1[7] = 0;

    i32x4 z4 = {0, 0, 0, 0};
    i32x8 z8 = {0, 0, 0, 0, 0, 0, 0, 0};
    __builtin_amdgcn_tensor_load_to_lds(g0, g1, z4, z4, z8, 0);
}
#endif

// ---------------------------------------------------------------------------
// GEMM: C[M,N] = (acc ? C : 0) + A[M,K]*W[N,K]^T + bias[N]; optional bf16 copy
// A, W pre-converted bf16. Block tile 64x64x64, 8 wave32 waves, each wave owns
// 16x32 of C -> 4 x v_wmma_f32_16x16x32_bf16 per K-tile. Tiles arrive via the
// Tensor Data Mover (double-buffered: tile i+1 in flight while i computes),
// synced with s_wait_tensorcnt + workgroup split-barriers.
// Requires K % 64 == 0, N % 64 == 0 (K in {768,256}, N=256 here).
// ---------------------------------------------------------------------------
#define BM 64
#define BN 64
#define BK 64

__global__ __launch_bounds__(256)
void gemm_bf16_wmma(const unsigned short* __restrict__ A, int M, int K,
                    const unsigned short* __restrict__ W, int N,
                    const float* __restrict__ bias,
                    float* __restrict__ C,
                    unsigned short* __restrict__ Cbf,
                    int accumulate)
{
    // [buffer][0=A,1=W][row][col] : 32 KB
    __shared__ alignas(16) unsigned short sT[2][2][BM][BK];

    const int tid  = threadIdx.x;
    const int lane = tid & 31;
    const int wave = tid >> 5;        // 0..7
    const int wy   = wave >> 1;       // 0..3 -> 16-row slice
    const int wx   = wave & 1;        // 0..1 -> 32-col half

    const int block_m = blockIdx.y * BM;
    const int block_n = blockIdx.x * BN;

    const int r  = lane & 15;
    const int kh = lane >> 4;         // 0/1

    v8f acc0 = {};
    v8f acc1 = {};

    const int ntiles = K / BK;

#ifdef HAS_TDM
    if (wave == 0) {
        tdm_load_tile_bf16(A, (unsigned)(uintptr_t)&sT[0][0][0][0],
                           block_m, M, 0, K, BM, BK);
        tdm_load_tile_bf16(W, (unsigned)(uintptr_t)&sT[0][1][0][0],
                           block_n, N, 0, K, BN, BK);
    }
#endif

    for (int i = 0; i < ntiles; ++i) {
        const int p = i & 1;
#ifdef HAS_TDM
        if (wave == 0) __builtin_amdgcn_s_wait_tensorcnt(0);
        __syncthreads();   // tile p visible; everyone done reading tile p^1
        if (wave == 0 && (i + 1) < ntiles) {
            const int k1 = (i + 1) * BK;
            tdm_load_tile_bf16(A, (unsigned)(uintptr_t)&sT[p ^ 1][0][0][0],
                               block_m, M, k1, K, BM, BK);
            tdm_load_tile_bf16(W, (unsigned)(uintptr_t)&sT[p ^ 1][1][0][0],
                               block_n, N, k1, K, BN, BK);
        }
#else
        __syncthreads();
        // synchronous cooperative fallback: 512 x 16B chunks per tile, 2/thread
        {
            const int k0 = i * BK;
            #pragma unroll
            for (int t = 0; t < 2; ++t) {
                const int c  = tid + t * 256;
                const int rr = c >> 3;
                const int cc = (c & 7) * 8;
                uint4 va = {0, 0, 0, 0}, vw = {0, 0, 0, 0};
                const int gm = block_m + rr;
                const int gn = block_n + rr;
                if (gm < M) va = *(const uint4*)(A + (size_t)gm * K + k0 + cc);
                if (gn < N) vw = *(const uint4*)(W + (size_t)gn * K + k0 + cc);
                *(uint4*)&sT[p][0][rr][cc] = va;
                *(uint4*)&sT[p][1][rr][cc] = vw;
            }
        }
        __syncthreads();
#endif
        // ---- two K=32 steps -> 4 WMMAs per wave per tile ----
        #pragma unroll
        for (int s = 0; s < 2; ++s) {
            FragU fa, fb0, fb1;
            {
                const int m = wy * 16 + r;
                fa.q[0] = *(const uint4*)&sT[p][0][m][s * 32 + kh * 16];
                fa.q[1] = *(const uint4*)&sT[p][0][m][s * 32 + kh * 16 + 8];
            }
            {
                const int n0 = wx * 32 + r;
                const int n1 = n0 + 16;
                fb0.q[0] = *(const uint4*)&sT[p][1][n0][s * 32 + kh * 16];
                fb0.q[1] = *(const uint4*)&sT[p][1][n0][s * 32 + kh * 16 + 8];
                fb1.q[0] = *(const uint4*)&sT[p][1][n1][s * 32 + kh * 16];
                fb1.q[1] = *(const uint4*)&sT[p][1][n1][s * 32 + kh * 16 + 8];
            }
            acc0 = __builtin_amdgcn_wmma_f32_16x16x32_bf16(
                       false, fa.v, false, fb0.v, (short)0, acc0, false, false);
            acc1 = __builtin_amdgcn_wmma_f32_16x16x32_bf16(
                       false, fa.v, false, fb1.v, (short)0, acc1, false, false);
        }
    }

    // ---- epilogue: C/D layout -> lane = N, vgpr idx (+8 hi half) = M ----
    const int cn0 = block_n + wx * 32 + (lane & 15);
    const int cn1 = cn0 + 16;
    const int mb  = block_m + wy * 16 + (kh ? 8 : 0);
    const float b0 = bias ? bias[cn0] : 0.0f;
    const float b1 = bias ? bias[cn1] : 0.0f;
    #pragma unroll
    for (int rr = 0; rr < 8; ++rr) {
        const int m = mb + rr;
        if (m < M) {
            float* c0 = C + (size_t)m * N + cn0;
            float* c1 = C + (size_t)m * N + cn1;
            float v0 = acc0[rr] + b0;
            float v1 = acc1[rr] + b1;
            if (accumulate) { v0 += *c0; v1 += *c1; }
            *c0 = v0;
            *c1 = v1;
            if (Cbf) {
                Cbf[(size_t)m * N + cn0] = f32_to_bf16_rne(v0);
                Cbf[(size_t)m * N + cn1] = f32_to_bf16_rne(v1);
            }
        }
    }
}

// ---------------------------------------------------------------------------
// Small projections (K=8 and K=1): memory-bound, plain VALU
// ---------------------------------------------------------------------------
__global__ __launch_bounds__(256)
void proj_url(const float* __restrict__ x, const float* __restrict__ Wu,
              const float* __restrict__ bu, float* __restrict__ out)
{
    __shared__ float xs[8];
    const int n = blockIdx.x;
    if (threadIdx.x < 8) xs[threadIdx.x] = x[(size_t)n * 8 + threadIdx.x];
    __syncthreads();
    const int h = threadIdx.x;
    float acc = bu[h];
    #pragma unroll
    for (int k = 0; k < 8; ++k) acc += xs[k] * Wu[h * 8 + k];
    out[(size_t)n * H + h] = acc;
}

__global__ __launch_bounds__(256)
void proj_sender(const float* __restrict__ x, const float* __restrict__ Ws,
                 const float* __restrict__ bs, float* __restrict__ out)
{
    const int n = blockIdx.x;
    const int h = threadIdx.x;
    out[(size_t)n * H + h] = x[n] * Ws[h] + bs[h];
}

// ---------------------------------------------------------------------------
// Edge scatter-add (segment sum) + count; 64 lanes per edge, float4 per lane
// ---------------------------------------------------------------------------
__global__ __launch_bounds__(256)
void scatter_add(const float* __restrict__ feat, const int* __restrict__ src,
                 const int* __restrict__ dst, int nE,
                 float* __restrict__ aggr, float* __restrict__ cnt)
{
    const int e = blockIdx.x * 4 + (threadIdx.x >> 6);
    if (e >= nE) return;
    const int l = threadIdx.x & 63;
    const int s = src[e];
    const int d = dst[e];
    const float4 v = *(const float4*)(feat + (size_t)s * H + l * 4);
    float* base = aggr + (size_t)d * H + l * 4;
    atomicAdd(base + 0, v.x);
    atomicAdd(base + 1, v.y);
    atomicAdd(base + 2, v.z);
    atomicAdd(base + 3, v.w);
    if (l == 0) atomicAdd(cnt + d, 1.0f);
}

// mean-normalize and emit bf16 (GEMM operand) in one pass
__global__ __launch_bounds__(256)
void normalize_bf16(const float* __restrict__ aggr, const float* __restrict__ cnt,
                    unsigned short* __restrict__ out, size_t total)
{
    size_t i = (size_t)blockIdx.x * blockDim.x + threadIdx.x;
    if (i < total) {
        const float c = fmaxf(cnt[i >> 8], 1.0f);
        out[i] = f32_to_bf16_rne(aggr[i] / c);
    }
}

// ---------------------------------------------------------------------------
// Fused: leaky-relu(0.5*(e1+e2)) -> gated residual -> KAN (SiLU base +
// Cox-de-Boor cubic B-splines on uniform grid [-2.2,2.2], h=0.4, 8 bases)
// ---------------------------------------------------------------------------
__global__ __launch_bounds__(256)
void combine_kan(const float* __restrict__ sum_e, const float* __restrict__ he_res,
                 const float* __restrict__ gate,
                 const float* __restrict__ base_w,     // [2][256]
                 const float* __restrict__ spline_w,   // [2][256][8]
                 float* __restrict__ out)              // [NE][2]
{
    const int n = blockIdx.x;
    const int i = threadIdx.x;

    float v = 0.5f * sum_e[(size_t)n * H + i];
    v = (v > 0.0f) ? v : 0.2f * v;                        // leaky relu 0.2
    const float alpha = 1.0f / (1.0f + __expf(-gate[0])); // sigmoid(gate)
    const float h = alpha * v + (1.0f - alpha) * he_res[(size_t)n * H + i];

    const float sil = h / (1.0f + __expf(-h));            // silu

    float bb[11];
    #pragma unroll
    for (int j = 0; j < 11; ++j) {
        const float gj = -2.2f + 0.4f * j;
        bb[j] = (h >= gj && h < gj + 0.4f) ? 1.0f : 0.0f;
    }
    #pragma unroll
    for (int p = 1; p <= 3; ++p) {
        const float inv = 1.0f / (0.4f * p);
        #pragma unroll
        for (int j = 0; j <= 10 - p; ++j) {
            const float gj = -2.2f + 0.4f * j;
            const float left  = (h - gj) * inv;
            const float right = ((gj + 0.4f * (p + 1)) - h) * inv;
            bb[j] = left * bb[j] + right * bb[j + 1];
        }
    }

    float o0 = sil * base_w[i];
    float o1 = sil * base_w[H + i];
    #pragma unroll
    for (int g = 0; g < 8; ++g) {
        o0 += bb[g] * spline_w[(size_t)i * 8 + g];
        o1 += bb[g] * spline_w[(size_t)(H + i) * 8 + g];
    }

    __shared__ float red0[256];
    __shared__ float red1[256];
    red0[i] = o0;
    red1[i] = o1;
    __syncthreads();
    #pragma unroll
    for (int s = 128; s > 0; s >>= 1) {
        if (i < s) { red0[i] += red0[i + s]; red1[i] += red1[i + s]; }
        __syncthreads();
    }
    if (i == 0) {
        out[(size_t)n * 2 + 0] = red0[0];
        out[(size_t)n * 2 + 1] = red1[0];
    }
}

// ---------------------------------------------------------------------------
// launch
// ---------------------------------------------------------------------------
extern "C" void kernel_launch(void* const* d_in, const int* in_sizes, int n_in,
                              void* d_out, int out_size, void* d_ws, size_t ws_size,
                              hipStream_t stream)
{
    const float* x_email      = (const float*)d_in[0];
    const float* x_url        = (const float*)d_in[1];
    const float* x_sender     = (const float*)d_in[2];
    const float* W_e          = (const float*)d_in[3];
    const float* b_e          = (const float*)d_in[4];
    const float* W_u          = (const float*)d_in[5];
    const float* b_u          = (const float*)d_in[6];
    const float* W_s          = (const float*)d_in[7];
    const float* b_s          = (const float*)d_in[8];
    // 9..11 (eu) dead — output depends only on the email branch
    const float* Wl_ue        = (const float*)d_in[12];
    const float* bl_ue        = (const float*)d_in[13];
    const float* Wr_ue        = (const float*)d_in[14];
    const float* Wl_se        = (const float*)d_in[15];
    const float* bl_se        = (const float*)d_in[16];
    const float* Wr_se        = (const float*)d_in[17];
    // 18..20 (es) dead
    const float* gate         = (const float*)d_in[21];
    const float* kan_base_w   = (const float*)d_in[22];
    const float* kan_spline_w = (const float*)d_in[23];
    // 24,25 eu edges dead
    const int*   ue_src       = (const int*)d_in[26];
    const int*   ue_dst       = (const int*)d_in[27];
    const int*   se_src       = (const int*)d_in[28];
    const int*   se_dst       = (const int*)d_in[29];
    // 30,31 es edges dead

    float* out = (float*)d_out;

    // ---- workspace layout ----
    float* ws    = (float*)d_ws;
    float* he    = ws;                              // [NE,H] f32 (residual)
    float* hu    = he    + (size_t)NE * H;          // [NU,H]
    float* hs    = hu    + (size_t)NU * H;          // [NS,H]
    float* aggr1 = hs    + (size_t)NS * H;          // [NE,H] sum(hu over ue)
    float* aggr2 = aggr1 + (size_t)NE * H;          // [NE,H] sum(hs over se)
    float* sum_e = aggr2 + (size_t)NE * H;          // [NE,H] out_e1+out_e2
    float* cnt1  = sum_e + (size_t)NE * H;          // [NE]
    float* cnt2  = cnt1  + NE;                      // [NE]
    unsigned short* bfbase = (unsigned short*)(cnt2 + NE);
    unsigned short* xe_bf  = bfbase;                           // [NE,768]
    unsigned short* he_bf  = xe_bf  + (size_t)NE * 768;        // [NE,H]
    unsigned short* a1_bf  = he_bf  + (size_t)NE * H;          // [NE,H]
    unsigned short* a2_bf  = a1_bf  + (size_t)NE * H;          // [NE,H]
    unsigned short* we_bf  = a2_bf  + (size_t)NE * H;          // [H,768]
    unsigned short* wlue_bf = we_bf  + (size_t)H * 768;        // [H,H] x4
    unsigned short* wrue_bf = wlue_bf + (size_t)H * H;
    unsigned short* wlse_bf = wrue_bf + (size_t)H * H;
    unsigned short* wrse_bf = wlse_bf + (size_t)H * H;

    // zero aggregation buffers + counts
    zero_f32<<<2048, 256, 0, stream>>>(aggr1, (size_t)2 * NE * H);
    zero_f32<<<256, 256, 0, stream>>>(cnt1, (size_t)2 * NE);

    // bf16 operand preparation (once per call; halves GEMM HBM traffic)
    cvt_f32_bf16<<<2048, 256, 0, stream>>>(x_email, xe_bf, (size_t)NE * 768);
    cvt_f32_bf16<<<192, 256, 0, stream>>>(W_e, we_bf, (size_t)H * 768);
    cvt_f32_bf16<<<64, 256, 0, stream>>>(Wl_ue, wlue_bf, (size_t)H * H);
    cvt_f32_bf16<<<64, 256, 0, stream>>>(Wr_ue, wrue_bf, (size_t)H * H);
    cvt_f32_bf16<<<64, 256, 0, stream>>>(Wl_se, wlse_bf, (size_t)H * H);
    cvt_f32_bf16<<<64, 256, 0, stream>>>(Wr_se, wrse_bf, (size_t)H * H);

    const dim3 gE(H / BN, (NE + BM - 1) / BM);

    // he = x_email @ W_e^T + b_e  (WMMA + TDM); emits f32 + bf16 copies
    gemm_bf16_wmma<<<gE, 256, 0, stream>>>(xe_bf, NE, 768, we_bf, H, b_e,
                                           he, he_bf, 0);
    proj_url<<<NU, 256, 0, stream>>>(x_url, W_u, b_u, hu);
    proj_sender<<<NS, 256, 0, stream>>>(x_sender, W_s, b_s, hs);

    // segment means onto email nodes
    scatter_add<<<(N_UE + 3) / 4, 256, 0, stream>>>(hu, ue_src, ue_dst, N_UE, aggr1, cnt1);
    scatter_add<<<(N_SE + 3) / 4, 256, 0, stream>>>(hs, se_src, se_dst, N_SE, aggr2, cnt2);
    {
        size_t tot = (size_t)NE * H;
        int blocks = (int)((tot + 255) / 256);
        normalize_bf16<<<blocks, 256, 0, stream>>>(aggr1, cnt1, a1_bf, tot);
        normalize_bf16<<<blocks, 256, 0, stream>>>(aggr2, cnt2, a2_bf, tot);
    }

    // sum_e = a1@Wl_ue^T + bl_ue + he@Wr_ue^T + a2@Wl_se^T + bl_se + he@Wr_se^T
    gemm_bf16_wmma<<<gE, 256, 0, stream>>>(a1_bf, NE, H, wlue_bf, H, bl_ue, sum_e, nullptr, 0);
    gemm_bf16_wmma<<<gE, 256, 0, stream>>>(he_bf, NE, H, wrue_bf, H, nullptr, sum_e, nullptr, 1);
    gemm_bf16_wmma<<<gE, 256, 0, stream>>>(a2_bf, NE, H, wlse_bf, H, bl_se, sum_e, nullptr, 1);
    gemm_bf16_wmma<<<gE, 256, 0, stream>>>(he_bf, NE, H, wrse_bf, H, nullptr, sum_e, nullptr, 1);

    // fused leaky-relu + gated residual + KAN head
    combine_kan<<<NE, 256, 0, stream>>>(sum_e, he, gate, kan_base_w, kan_spline_w, out);
}
